// MultiHeadAttention_16707422782054
// MI455X (gfx1250) — compile-verified, matrix operand order checked by reading, not run on hardware
//
#include <hip/hip_runtime.h>
#include <hip/hip_bf16.h>

// ---------------------------------------------------------------------------
// MHA forward for MI455X (gfx1250, wave32, WMMA).
// Compute-bound on matrix ops (~69 GFLOP vs ~83MB traffic @23.3TB/s): all
// GEMM-like work goes through v_wmma_f32_16x16x32_bf16 with f32 accumulation.
// Softmax cross-lane reductions use DPP (VALU). V-chunk staging uses the
// CDNA5 async-to-LDS DMA path (GLOBAL_LOAD_ASYNC_TO_LDS_B128 + ASYNCcnt)
// when the toolchain exposes it; software transpose staging as fallback.
// ---------------------------------------------------------------------------

typedef __attribute__((ext_vector_type(16))) __bf16 v16bf;
typedef __attribute__((ext_vector_type(8)))  __bf16 bf16x8;
typedef __attribute__((ext_vector_type(8)))  float  v8f;

// Exact pointee type of the async-LDS builtin (from the compiler diagnostic):
// 'int __attribute__((vector_size(16)))' in address spaces 1 (global) / 3 (LDS).
typedef int v4i_vs __attribute__((vector_size(16)));
typedef __attribute__((address_space(1))) v4i_vs gv4i;
typedef __attribute__((address_space(3))) v4i_vs lv4i;

#if defined(__HIP_DEVICE_COMPILE__)
#if __has_builtin(__builtin_amdgcn_global_load_async_to_lds_b128)
#define HAVE_ASYNC_LDS 1
#endif
#endif

#define SEQ 2048
#define DM  1024   // H * DK
#define HH  16
#define DKK 64

static __device__ __forceinline__ v8f vzero8() {
  v8f z = {0.f, 0.f, 0.f, 0.f, 0.f, 0.f, 0.f, 0.f};
  return z;
}

// Build a 16x32 (A) / 32x16 (B) bf16 fragment for one lane from two aligned
// 8-element runs (ISA 7.12.2: lane holds K=[kb,kb+8) and K=[16+kb,16+kb+8)).
static __device__ __forceinline__ v16bf load_frag(const __bf16* lo, const __bf16* hi) {
  bf16x8 a = *(const bf16x8*)lo;
  bf16x8 b = *(const bf16x8*)hi;
  v16bf f;
#pragma unroll
  for (int i = 0; i < 8; ++i) { f[i] = a[i]; f[i + 8] = b[i]; }
  return f;
}

// ---------------------------------------------------------------------------
// Row-group (16-lane) reductions via DPP butterfly (VALU, no LDS traffic).
// ---------------------------------------------------------------------------
#if __has_builtin(__builtin_amdgcn_mov_dpp)
template <int CTRL>
static __device__ __forceinline__ float dpp_f32(float x) {
  return __int_as_float(
      __builtin_amdgcn_mov_dpp(__float_as_int(x), CTRL, 0xF, 0xF, true));
}
static __device__ __forceinline__ float rowmax16(float v) {
  v = fmaxf(v, dpp_f32<0xB1>(v));    // quad_perm [1,0,3,2]
  v = fmaxf(v, dpp_f32<0x4E>(v));    // quad_perm [2,3,0,1]
  v = fmaxf(v, dpp_f32<0x141>(v));   // row_half_mirror
  v = fmaxf(v, dpp_f32<0x140>(v));   // row_mirror
  return v;
}
static __device__ __forceinline__ float rowsum16(float v) {
  v += dpp_f32<0xB1>(v);
  v += dpp_f32<0x4E>(v);
  v += dpp_f32<0x141>(v);
  v += dpp_f32<0x140>(v);
  return v;
}
#else
static __device__ __forceinline__ float rowmax16(float v) {
#pragma unroll
  for (int m = 1; m < 16; m <<= 1) v = fmaxf(v, __shfl_xor(v, m, 32));
  return v;
}
static __device__ __forceinline__ float rowsum16(float v) {
#pragma unroll
  for (int m = 1; m < 16; m <<= 1) v += __shfl_xor(v, m, 32);
  return v;
}
#endif

// ---------------------------------------------------------------------------
// GEMM: out[M,N] = A[M,K] @ W[K,N] + bias.  128 threads = 4 waves.
// WG tile 64x64, K-step 64: 2 barriers amortized over 8 WMMAs per wave.
// ---------------------------------------------------------------------------
template <typename AT, typename OT>
__global__ void __launch_bounds__(128)
gemm_bias_wmma(const AT* __restrict__ A, const float* __restrict__ W,
               const float* __restrict__ bias, OT* __restrict__ out,
               int M, int N, int K) {
  __shared__ __bf16 sA[64][64];   // [m][k]
  __shared__ __bf16 sBt[64][64];  // [n][k] (W transposed)
  const int t    = threadIdx.x;
  const int lane = t & 31;
  const int wave = t >> 5;
  const int lm   = lane & 15;
  const int kb   = (lane < 16) ? 0 : 8;
  const int off  = (lane < 16) ? 0 : 8;
  const int mBase = blockIdx.x * 64;
  const int nBase = blockIdx.y * 64;

  v8f acc[4] = {vzero8(), vzero8(), vzero8(), vzero8()};

  for (int k0 = 0; k0 < K; k0 += 64) {
    __syncthreads();
    // Stage A tile 64x64 -> bf16 LDS (2 threads per row, 32 elems each).
    {
      const int i = t >> 1, jh = (t & 1) * 32;
      const AT* src = A + (size_t)(mBase + i) * K + k0 + jh;
#pragma unroll
      for (int j = 0; j < 32; ++j) sA[i][jh + j] = (__bf16)(float)src[j];
      if (k0 + 64 < K) __builtin_prefetch(src + 64, 0, 1);
    }
    // Stage W tile 64x64 -> transposed bf16 LDS.
    {
      const int kk = t & 63, nh = t >> 6;
      const float* src = W + (size_t)(k0 + kk) * N + nBase + nh * 32;
#pragma unroll
      for (int j = 0; j < 32; ++j) sBt[nh * 32 + j][kk] = (__bf16)src[j];
      if (k0 + 64 < K) __builtin_prefetch(src + (size_t)64 * N, 0, 1);
    }
    __syncthreads();

    const int mrow = wave * 16 + lm;
    v16bf af[2];
#pragma unroll
    for (int dc = 0; dc < 2; ++dc)
      af[dc] = load_frag(&sA[mrow][dc * 32 + kb], &sA[mrow][dc * 32 + 16 + kb]);
#pragma unroll
    for (int n = 0; n < 4; ++n) {
      const int nrow = n * 16 + lm;
      v16bf b0 = load_frag(&sBt[nrow][kb],      &sBt[nrow][16 + kb]);
      v16bf b1 = load_frag(&sBt[nrow][32 + kb], &sBt[nrow][48 + kb]);
      acc[n] = __builtin_amdgcn_wmma_f32_16x16x32_bf16(
          false, af[0], false, b0, (short)0, acc[n], false, false);
      acc[n] = __builtin_amdgcn_wmma_f32_16x16x32_bf16(
          false, af[1], false, b1, (short)0, acc[n], false, false);
    }
  }

  // Epilogue: C/D layout -> lane holds rows r+off, column lm (ISA 7.12.2).
#pragma unroll
  for (int n = 0; n < 4; ++n) {
    const int col = nBase + n * 16 + lm;
    const float bv = bias[col];
#pragma unroll
    for (int r = 0; r < 8; ++r) {
      const int row = mBase + wave * 16 + r + off;
      out[(size_t)row * N + col] = (OT)(acc[n][r] + bv);
    }
  }
}

// ---------------------------------------------------------------------------
// Flash-style attention. Grid: (S/64, B*H). 128 threads = 4 waves; each wave
// owns a 16-query tile; the WG shares one 32-key V chunk staged in LDS.
// ---------------------------------------------------------------------------
__global__ void __launch_bounds__(128)
mha_attention_wmma(const __bf16* __restrict__ qw, const __bf16* __restrict__ kw,
                   const __bf16* __restrict__ vw, __bf16* __restrict__ attn) {
#if HAVE_ASYNC_LDS
  __shared__ __bf16 sV[32][64];      // [key][dv], filled by async DMA
#else
  __shared__ __bf16 sVt[64][32];     // [dv][key], software transpose
#endif
  __shared__ __bf16 sP[4][16][32];   // per-wave P tile (16 queries x 32 keys)
  const int t    = threadIdx.x;
  const int lane = t & 31;
  const int wave = t >> 5;
  const int lm   = lane & 15;
  const int kb   = (lane < 16) ? 0 : 8;
  const int off  = (lane < 16) ? 0 : 8;
  const int b    = blockIdx.y / HH;
  const int h    = blockIdx.y % HH;
  const int qStart = blockIdx.x * 64 + wave * 16;
  const size_t baseRow = (size_t)b * SEQ;
  const int hcol = h * DKK;

  // Q fragments (16 queries x 64 dk as two 16x32 fragments), direct from
  // global bf16: each lane's data is two contiguous 16B runs of its row.
  v16bf qf[2];
  {
    const __bf16* qrow = qw + (baseRow + qStart + lm) * DM + hcol;
#pragma unroll
    for (int dc = 0; dc < 2; ++dc)
      qf[dc] = load_frag(qrow + dc * 32 + kb, qrow + dc * 32 + 16 + kb);
  }

  v8f o[4] = {vzero8(), vzero8(), vzero8(), vzero8()};
  float mr[8], lr[8];
#pragma unroll
  for (int r = 0; r < 8; ++r) { mr[r] = -1.0e30f; lr[r] = 0.f; }

  for (int c = 0; c < SEQ / 32; ++c) {
    const int kBase = c * 32;
    __syncthreads();

#if HAVE_ASYNC_LDS
    // Async DMA the V chunk (32 keys x 64 dv bf16 = 4KB) row-major into LDS:
    // 256 x 16B segments over 128 threads, tracked with ASYNCcnt.
#pragma unroll
    for (int it = 0; it < 2; ++it) {
      const int idx = t + it * 128;
      const int row = idx >> 3;
      const int sg  = (idx & 7) * 8;
      const __bf16* g = vw + (baseRow + kBase + row) * DM + hcol + sg;
      __builtin_amdgcn_global_load_async_to_lds_b128(
          (gv4i*)(void*)g, (lv4i*)&sV[row][sg], 0, 0);
    }
#if __has_builtin(__builtin_amdgcn_s_wait_asynccnt)
    __builtin_amdgcn_s_wait_asynccnt(0);
#else
    asm volatile("s_wait_asynccnt 0" ::: "memory");
#endif
#else
    // Software path: stage V chunk transposed via VGPR round-trip.
    {
      const int kk = t & 31, nh = t >> 5;
      const __bf16* src = vw + (baseRow + kBase + kk) * DM + hcol + nh * 16;
#pragma unroll
      for (int j = 0; j < 16; ++j) sVt[nh * 16 + j][kk] = src[j];
    }
#endif
    __syncthreads();

    // Preload all K fragments (one load clause), then 4 back-to-back WMMAs.
    v16bf kf[4];
#pragma unroll
    for (int st = 0; st < 2; ++st) {
      const __bf16* krow = kw + (baseRow + kBase + st * 16 + lm) * DM + hcol;
#pragma unroll
      for (int dc = 0; dc < 2; ++dc)
        kf[st * 2 + dc] =
            load_frag(krow + dc * 32 + kb, krow + dc * 32 + 16 + kb);
    }
    v8f s[2] = {vzero8(), vzero8()};
#pragma unroll
    for (int dc = 0; dc < 2; ++dc) {
      s[0] = __builtin_amdgcn_wmma_f32_16x16x32_bf16(
          false, qf[dc], false, kf[dc], (short)0, s[0], false, false);
      s[1] = __builtin_amdgcn_wmma_f32_16x16x32_bf16(
          false, qf[dc], false, kf[2 + dc], (short)0, s[1], false, false);
    }

    // Online softmax over the 32 new keys (reductions within 16-lane halves).
#pragma unroll
    for (int r = 0; r < 8; ++r) {
      float s0 = s[0][r] * 0.125f;   // 1/sqrt(64)
      float s1 = s[1][r] * 0.125f;
      const float mx    = rowmax16(fmaxf(s0, s1));
      const float mnew  = fmaxf(mr[r], mx);
      const float alpha = __expf(mr[r] - mnew);
      const float p0 = __expf(s0 - mnew);
      const float p1 = __expf(s1 - mnew);
      const float ps = rowsum16(p0 + p1);
      lr[r] = lr[r] * alpha + ps;
      mr[r] = mnew;
#pragma unroll
      for (int n = 0; n < 4; ++n) o[n][r] *= alpha;
      // Scatter P (C-layout) into LDS so it can be regathered in A-layout.
      sP[wave][r + off][lm]      = (__bf16)p0;
      sP[wave][r + off][16 + lm] = (__bf16)p1;
    }
    __syncthreads();

    // P (16x32) @ V (32x64): preload P + 4 V fragments, then 4 WMMAs.
    v16bf pf = load_frag(&sP[wave][lm][kb], &sP[wave][lm][16 + kb]);
    v16bf vf[4];
#pragma unroll
    for (int n = 0; n < 4; ++n) {
#if HAVE_ASYNC_LDS
      // Strided gather of the lane's dv column from row-major sV.
      const __bf16* col = &sV[0][n * 16 + lm];
#pragma unroll
      for (int j = 0; j < 8; ++j) vf[n][j] = col[(kb + j) * 64];
#pragma unroll
      for (int j = 0; j < 8; ++j) vf[n][8 + j] = col[(16 + kb + j) * 64];
#else
      vf[n] = load_frag(&sVt[n * 16 + lm][kb], &sVt[n * 16 + lm][16 + kb]);
#endif
    }
#pragma unroll
    for (int n = 0; n < 4; ++n)
      o[n] = __builtin_amdgcn_wmma_f32_16x16x32_bf16(
          false, pf, false, vf[n], (short)0, o[n], false, false);

    // Prefetch next V chunk (lowers to global_prefetch_b8).
    if (c + 1 < SEQ / 32) {
      const __bf16* nxt = vw + (baseRow + kBase + 32 + (t & 31)) * DM + hcol;
      __builtin_prefetch(nxt, 0, 1);
    }
  }

  // Normalize and store merged-head output [B,S,H*DV] as bf16.
#pragma unroll
  for (int n = 0; n < 4; ++n) {
#pragma unroll
    for (int r = 0; r < 8; ++r) {
      const int row = qStart + r + off;
      attn[(baseRow + row) * DM + hcol + n * 16 + lm] = (__bf16)(o[n][r] / lr[r]);
    }
  }
}

// ---------------------------------------------------------------------------
extern "C" void kernel_launch(void* const* d_in, const int* in_sizes, int n_in,
                              void* d_out, int out_size, void* d_ws, size_t ws_size,
                              hipStream_t stream) {
  (void)in_sizes; (void)n_in; (void)out_size; (void)ws_size;
  const float* q  = (const float*)d_in[0];
  const float* k  = (const float*)d_in[1];
  const float* v  = (const float*)d_in[2];
  const float* Wq = (const float*)d_in[3];
  const float* bq = (const float*)d_in[4];
  const float* Wk = (const float*)d_in[5];
  const float* bk = (const float*)d_in[6];
  const float* Wv = (const float*)d_in[7];
  const float* bv = (const float*)d_in[8];
  const float* Wo = (const float*)d_in[9];
  const float* bo = (const float*)d_in[10];
  float* out = (float*)d_out;

  const int B = 2, S = SEQ, D = DM;
  const int M = B * S;                       // 4096
  const size_t fsz = (size_t)M * D;          // elements per bf16 field

  char* ws = (char*)d_ws;
  __bf16* qwv = (__bf16*)(ws);
  __bf16* kwv = (__bf16*)(ws + fsz * 2);
  __bf16* vwv = (__bf16*)(ws + fsz * 4);
  __bf16* aw  = (__bf16*)(ws + fsz * 6);     // 32 MB total

  dim3 blk(128);
  dim3 gg(M / 64, D / 64);                   // 64 x 16 WGs per GEMM
  gemm_bias_wmma<float, __bf16><<<gg, blk, 0, stream>>>(q, Wq, bq, qwv, M, D, D);
  gemm_bias_wmma<float, __bf16><<<gg, blk, 0, stream>>>(k, Wk, bk, kwv, M, D, D);
  gemm_bias_wmma<float, __bf16><<<gg, blk, 0, stream>>>(v, Wv, bv, vwv, M, D, D);

  dim3 ga(S / 64, B * HH);                   // 32 x 32 WGs
  mha_attention_wmma<<<ga, blk, 0, stream>>>(qwv, kwv, vwv, aw);

  gemm_bias_wmma<__bf16, float><<<gg, blk, 0, stream>>>(aw, Wo, bo, out, M, D, D);
}
